// GCN_68719476736452
// MI455X (gfx1250) — matrix operand order repaired and checked by reading, not compiled, not run on hardware
//
#include <hip/hip_runtime.h>
#include <hip/hip_bf16.h>

// ---------------------------------------------------------------------------
// GCN forward for MI455X (gfx1250, wave32, WMMA + TDM).
//   N=100000 nodes, E=640000 edges, D=128, G=64 graphs, L=3 GCN layers.
// GEMMs use v_wmma_f32_16x16x32_bf16 (bf16 A from activations, bf16 W^T
// staged into LDS via the Tensor Data Mover when available), scatter uses
// global f32 atomics, BN stats use LDS f32 atomics + global atomics.
// conv_b cancels inside train-mode BN and is skipped on purpose.
// ---------------------------------------------------------------------------

#define DCH 128

typedef __attribute__((ext_vector_type(16))) __bf16 bf16x16;
typedef __attribute__((ext_vector_type(4)))  __bf16 bf16x4;
typedef __attribute__((ext_vector_type(8)))  float  f32x8;
typedef __attribute__((ext_vector_type(4)))  unsigned int u32x4v;
typedef __attribute__((ext_vector_type(8)))  int i32x8v;
typedef __attribute__((ext_vector_type(4)))  int i32x4v;

union FragBF16 { bf16x16 v; uint4 q[2]; };

#if __has_builtin(__builtin_amdgcn_tensor_load_to_lds) && \
    __has_builtin(__builtin_amdgcn_s_wait_tensorcnt)
#define USE_TDM 1
#else
#define USE_TDM 0
#endif

// ---------------------------------------------------------------------------
// WMMA GEMM:  C[nrows x 128] = A[nrows x 128](bf16) * W[128 x 128] (+ bias)
// WT is W transposed, bf16: WT[j*128 + k] = W[k][j].
// One wave computes a 16-row strip across all 128 output columns.
// W^T (32 KB) is staged into LDS by the TDM (one descriptor, one wave),
// tracked with TENSORcnt; workgroup barrier publishes it to all 8 waves.
// ---------------------------------------------------------------------------
template<bool STATS, bool WB16>
__global__ void __launch_bounds__(256)
k_gemm_wmma(const __bf16* __restrict__ A, const __bf16* __restrict__ WT,
            const float* __restrict__ bias, float* __restrict__ C,
            __bf16* __restrict__ Cb, float* __restrict__ stats, int nrows)
{
    __shared__ __bf16 ldsWT[DCH * DCH];           // 32 KB
#if USE_TDM
    if (threadIdx.x < 32) {   // TDM is a per-wave DMA op; one wave issues it
        unsigned long long ga = (unsigned long long)WT;
        unsigned lds_base = (unsigned)(unsigned long long)(void*)ldsWT;
        // D# group 0: count=1 | lds_addr | global_addr[56:0] | type=2
        u32x4v g0;
        g0[0] = 1u;
        g0[1] = lds_base;
        g0[2] = (unsigned)ga;
        g0[3] = (unsigned)((ga >> 32) & 0x01FFFFFFull) | (2u << 30);
        // D# group 1: data_size=2B; tensor 128x128; tile 128x128; stride 128
        i32x8v g1;
        g1[0] = 0x00010000;        // workgroup_mask=0, data_size=1 (2 bytes)
        g1[1] = 128 << 16;         // tensor_dim0[15:0] in bits 63:48
        g1[2] = 128 << 16;         // tensor_dim1[15:0] in bits 95:80
        g1[3] = 128 << 16;         // tile_dim0=128 in bits 127:112
        g1[4] = 128;               // tile_dim1=128; tile_dim2=0
        g1[5] = 128;               // tensor_dim0_stride = 128 elements
        g1[6] = 0;
        g1[7] = 0;
        i32x4v gz4 = {0, 0, 0, 0};                  // groups 2/3 unused (2-D)
        i32x8v gz8 = {0, 0, 0, 0, 0, 0, 0, 0};
        __builtin_amdgcn_tensor_load_to_lds(g0, g1, gz4, gz4, gz8, 0);
        __builtin_amdgcn_s_wait_tensorcnt(0);
    }
#else
    {   // cooperative stage of W^T into LDS (b128 loads / ds b128 stores)
        const uint4* s = (const uint4*)WT;
        uint4*       d = (uint4*)ldsWT;
        for (int i = threadIdx.x; i < DCH * DCH * 2 / 16; i += 256) d[i] = s[i];
    }
#endif
    __syncthreads();

    const int lane = threadIdx.x & 31;
    const int wv   = threadIdx.x >> 5;            // 8 waves / block
    const int m16  = lane & 15;
    const int hi   = lane >> 4;                   // lane-half selector
    const int nstrips = nrows >> 4;

    for (int strip = blockIdx.x * 8 + wv; strip < nstrips; strip += gridDim.x * 8) {
        const int row0 = strip << 4;

        // A fragments: 16x32 bf16 tile per K-step, ISA layout:
        // lanes 0-15: row=lane, elems 0-7 -> K=k0+0..7,  8-15 -> K=k0+16..23
        // lanes16-31: row=lane-16, elems 0-7 -> K=k0+8..15, 8-15 -> K=k0+24..31
        FragBF16 a[4];
        const __bf16* arow = A + (size_t)(row0 + m16) * DCH + hi * 8;
        #pragma unroll
        for (int k = 0; k < 4; ++k) {
            a[k].q[0] = *(const uint4*)(arow + k * 32);
            a[k].q[1] = *(const uint4*)(arow + k * 32 + 16);
        }

        #pragma unroll
        for (int nt = 0; nt < 8; ++nt) {
            const int col = nt * 16 + m16;        // output column for this lane
            const float bv = bias ? bias[col] : 0.0f;
            f32x8 acc;
            #pragma unroll
            for (int r = 0; r < 8; ++r) acc[r] = bv;

            // B fragments from LDS W^T: same K layout as A (K=k0+8h+0..7, k0+16+8h+0..7).
            const __bf16* bcol = ldsWT + col * DCH + hi * 8;
            #pragma unroll
            for (int k = 0; k < 4; ++k) {
                FragBF16 b;
                b.q[0] = *(const uint4*)(bcol + k * 32);
                b.q[1] = *(const uint4*)(bcol + k * 32 + 16);
                acc = __builtin_amdgcn_wmma_f32_16x16x32_bf16(
                          false, a[k].v, false, b.v, (short)0, acc, false, false);
            }

            // C layout: VGPR r -> rows (row0 + hi*8 + r), lane -> column.
            const int mbase = row0 + hi * 8;
            float lsum = 0.0f, lsq = 0.0f;
            #pragma unroll
            for (int r = 0; r < 8; ++r) {
                const float v = acc[r];
                C[(size_t)(mbase + r) * DCH + col] = v;
                if (WB16) Cb[(size_t)(mbase + r) * DCH + col] = (__bf16)v;
                if (STATS) { lsum += v; lsq += v * v; }
            }
            if (STATS) {
                unsafeAtomicAdd(&stats[col], lsum);
                unsafeAtomicAdd(&stats[DCH + col], lsq);
            }
        }
    }
}

// ---------------------------------------------------------------------------
// Degree / normalization
// ---------------------------------------------------------------------------
__global__ void __launch_bounds__(256)
k_deg(const int* __restrict__ dst, float* __restrict__ deg, int E)
{
    for (int e = blockIdx.x * 256 + threadIdx.x; e < E; e += gridDim.x * 256)
        unsafeAtomicAdd(&deg[dst[e]], 1.0f);
}

__global__ void __launch_bounds__(256)
k_dinv(float* __restrict__ deg, int n)
{
    int i = blockIdx.x * 256 + threadIdx.x;
    if (i < n) {
        float d = deg[i];
        deg[i] = (d > 0.5f) ? rsqrtf(d) : 0.0f;
    }
}

// ---------------------------------------------------------------------------
// Weight transpose + bf16 convert: WT[j*128+k] = (bf16)W[k*128+j]
// ---------------------------------------------------------------------------
__global__ void __launch_bounds__(256)
k_wt(const float* __restrict__ W, __bf16* __restrict__ WT)
{
    for (int i = blockIdx.x * 256 + threadIdx.x; i < DCH * DCH; i += gridDim.x * 256) {
        int k = i >> 7, j = i & 127;
        WT[j * DCH + k] = (__bf16)W[i];
    }
}

__global__ void __launch_bounds__(256)
k_bf16cast(const float* __restrict__ in, __bf16* __restrict__ out, int n4)
{
    for (int i = blockIdx.x * 256 + threadIdx.x; i < n4; i += gridDim.x * 256) {
        float4 v = ((const float4*)in)[i];
        bf16x4 o = { (__bf16)v.x, (__bf16)v.y, (__bf16)v.z, (__bf16)v.w };
        ((bf16x4*)out)[i] = o;
    }
}

// ---------------------------------------------------------------------------
// BN finalize: scale = g*rsqrt(var+eps); shift = beta - mean*scale
// ---------------------------------------------------------------------------
__global__ void k_finalize(const float* __restrict__ stats,
                           const float* __restrict__ g,
                           const float* __restrict__ beta,
                           float* __restrict__ scsh, float invN)
{
    int c = threadIdx.x;                           // 128 threads
    float mean = stats[c] * invN;
    float var  = stats[DCH + c] * invN - mean * mean;
    float inv  = rsqrtf(var + 1e-5f);
    float sc   = g[c] * inv;
    scsh[c]       = sc;
    scsh[DCH + c] = beta[c] - mean * sc;
}

// encoder: hb = bf16(relu(t*sc + sh))
__global__ void __launch_bounds__(256)
k_bnrelu(const float* __restrict__ t, const float* __restrict__ scsh,
         __bf16* __restrict__ out, int n4)
{
    for (int i = blockIdx.x * 256 + threadIdx.x; i < n4; i += gridDim.x * 256) {
        float4 v = ((const float4*)t)[i];
        int c = (i & 31) * 4;
        float a0 = fmaxf(fmaf(v.x, scsh[c + 0], scsh[DCH + c + 0]), 0.0f);
        float a1 = fmaxf(fmaf(v.y, scsh[c + 1], scsh[DCH + c + 1]), 0.0f);
        float a2 = fmaxf(fmaf(v.z, scsh[c + 2], scsh[DCH + c + 2]), 0.0f);
        float a3 = fmaxf(fmaf(v.w, scsh[c + 3], scsh[DCH + c + 3]), 0.0f);
        bf16x4 o = { (__bf16)a0, (__bf16)a1, (__bf16)a2, (__bf16)a3 };
        ((bf16x4*)out)[i] = o;
    }
}

// residual: h += relu(agg*sc + sh); refresh bf16 copy
__global__ void __launch_bounds__(256)
k_update(float* __restrict__ h, __bf16* __restrict__ hb,
         const float* __restrict__ agg, const float* __restrict__ scsh, int n4)
{
    for (int i = blockIdx.x * 256 + threadIdx.x; i < n4; i += gridDim.x * 256) {
        float4 hv = ((const float4*)h)[i];
        float4 av = ((const float4*)agg)[i];
        int c = (i & 31) * 4;
        hv.x += fmaxf(fmaf(av.x, scsh[c + 0], scsh[DCH + c + 0]), 0.0f);
        hv.y += fmaxf(fmaf(av.y, scsh[c + 1], scsh[DCH + c + 1]), 0.0f);
        hv.z += fmaxf(fmaf(av.z, scsh[c + 2], scsh[DCH + c + 2]), 0.0f);
        hv.w += fmaxf(fmaf(av.w, scsh[c + 3], scsh[DCH + c + 3]), 0.0f);
        ((float4*)h)[i] = hv;
        bf16x4 o = { (__bf16)hv.x, (__bf16)hv.y, (__bf16)hv.z, (__bf16)hv.w };
        ((bf16x4*)hb)[i] = o;
    }
}

// ---------------------------------------------------------------------------
// Edge scatter: agg[dst] += hw[src] * dinv[src]*dinv[dst]. One wave per edge.
// ---------------------------------------------------------------------------
__global__ void __launch_bounds__(256)
k_scatter(const float* __restrict__ hw, const int* __restrict__ src,
          const int* __restrict__ dst, const float* __restrict__ dinv,
          float* __restrict__ agg, int E)
{
    const int lane = threadIdx.x & 31;
    const int w    = (blockIdx.x * 256 + threadIdx.x) >> 5;
    const int nw   = (gridDim.x * 256) >> 5;
    for (int e = w; e < E; e += nw) {
        const int s = src[e], d = dst[e];
        const int e2 = e + nw;
        if (e2 < E) __builtin_prefetch(hw + (size_t)src[e2] * DCH, 0, 0);
        const float nrm = dinv[s] * dinv[d];
        float4 v = ((const float4*)(hw + (size_t)s * DCH))[lane];
        float* p = agg + (size_t)d * DCH + lane * 4;
        unsafeAtomicAdd(p + 0, v.x * nrm);
        unsafeAtomicAdd(p + 1, v.y * nrm);
        unsafeAtomicAdd(p + 2, v.z * nrm);
        unsafeAtomicAdd(p + 3, v.w * nrm);
    }
}

// ---------------------------------------------------------------------------
// Per-channel sum / sumsq over rows (LDS reduction then global atomics)
// ---------------------------------------------------------------------------
__global__ void __launch_bounds__(256)
k_colstats(const float* __restrict__ agg, float* __restrict__ stats, int nrows)
{
    __shared__ float ssum[DCH], ssq[DCH];
    const int tid = threadIdx.x;
    if (tid < DCH) { ssum[tid] = 0.0f; ssq[tid] = 0.0f; }
    __syncthreads();
    const int col = tid & 127, rg = tid >> 7;
    float s = 0.0f, s2 = 0.0f;
    for (int r = blockIdx.x * 2 + rg; r < nrows; r += gridDim.x * 2) {
        float v = agg[(size_t)r * DCH + col];
        s += v; s2 += v * v;
    }
    unsafeAtomicAdd(&ssum[col], s);      // ds_add_f32
    unsafeAtomicAdd(&ssq[col], s2);
    __syncthreads();
    if (tid < DCH) {
        unsafeAtomicAdd(&stats[col], ssum[col]);
        unsafeAtomicAdd(&stats[DCH + col], ssq[col]);
    }
}

// ---------------------------------------------------------------------------
// Graph pooling: hg[batch[n]] += h[n]
// ---------------------------------------------------------------------------
__global__ void __launch_bounds__(256)
k_pool(const float* __restrict__ h, const int* __restrict__ batch,
       float* __restrict__ hg, int n)
{
    const int total = n * 32;
    for (int i = blockIdx.x * 256 + threadIdx.x; i < total; i += gridDim.x * 256) {
        const int node = i >> 5, q = i & 31;
        float4 v = ((const float4*)(h + (size_t)node * DCH))[q];
        float* p = hg + batch[node] * DCH + q * 4;
        unsafeAtomicAdd(p + 0, v.x);
        unsafeAtomicAdd(p + 1, v.y);
        unsafeAtomicAdd(p + 2, v.z);
        unsafeAtomicAdd(p + 3, v.w);
    }
}

// ---------------------------------------------------------------------------
// MLP readout: 64x128 -> 64 -> 32 -> 1, single block, LDS-staged.
// ---------------------------------------------------------------------------
__global__ void __launch_bounds__(256)
k_readout(const float* __restrict__ hg,
          const float* __restrict__ W1, const float* __restrict__ b1,
          const float* __restrict__ W2, const float* __restrict__ b2,
          const float* __restrict__ W3, const float* __restrict__ b3,
          float* __restrict__ out)
{
    __shared__ float s_hg[64 * 128];
    __shared__ float s_y1[64 * 64];
    __shared__ float s_y2[64 * 32];
    const int tid = threadIdx.x;
    for (int i = tid; i < 64 * 128; i += 256) s_hg[i] = hg[i];
    __syncthreads();
    for (int i = tid; i < 64 * 64; i += 256) {
        int g = i >> 6, j = i & 63;
        float s = b1[j];
        for (int k = 0; k < 128; ++k) s = fmaf(s_hg[g * 128 + k], W1[k * 64 + j], s);
        s_y1[i] = fmaxf(s, 0.0f);
    }
    __syncthreads();
    for (int i = tid; i < 64 * 32; i += 256) {
        int g = i >> 5, j = i & 31;
        float s = b2[j];
        for (int k = 0; k < 64; ++k) s = fmaf(s_y1[g * 64 + k], W2[k * 32 + j], s);
        s_y2[i] = fmaxf(s, 0.0f);
    }
    __syncthreads();
    if (tid < 64) {
        float s = b3[0];
        for (int k = 0; k < 32; ++k) s = fmaf(s_y2[tid * 32 + k], W3[k], s);
        out[tid] = s;
    }
}

// ---------------------------------------------------------------------------
extern "C" void kernel_launch(void* const* d_in, const int* in_sizes, int n_in,
                              void* d_out, int out_size, void* d_ws, size_t ws_size,
                              hipStream_t stream)
{
    const float* x    = (const float*)d_in[0];
    const int*   ei   = (const int*)d_in[1];
    const int*   batc = (const int*)d_in[2];
    const float* eW1  = (const float*)d_in[3];
    const float* eb1  = (const float*)d_in[4];
    const float* eg   = (const float*)d_in[5];
    const float* ebe  = (const float*)d_in[6];
    const float* eW2  = (const float*)d_in[7];
    const float* eb2  = (const float*)d_in[8];
    const float* cW   = (const float*)d_in[9];
    // d_in[10] = conv_b: algebraically cancels inside train-mode BN (x - mean)
    const float* bng  = (const float*)d_in[11];
    const float* bnb  = (const float*)d_in[12];
    const float* rW1  = (const float*)d_in[13];
    const float* rb1  = (const float*)d_in[14];
    const float* rW2  = (const float*)d_in[15];
    const float* rb2  = (const float*)d_in[16];
    const float* rW3  = (const float*)d_in[17];
    const float* rb3  = (const float*)d_in[18];

    const int N = in_sizes[0] / DCH;      // 100000
    const int E = in_sizes[1] / 2;        // 640000
    const int n4 = N * (DCH / 4);         // float4 elements per activation

    char* w = (char*)d_ws;
    size_t off = 0;
    auto take = [&](size_t bytes) -> void* {
        void* p = w + off;
        off = (off + bytes + 255) & ~(size_t)255;
        return p;
    };
    float*   h     = (float*)take((size_t)N * DCH * 4);
    __bf16*  hb    = (__bf16*)take((size_t)N * DCH * 2);   // also x(bf16) at start
    float*   bufA  = (float*)take((size_t)N * DCH * 4);    // t / hw
    float*   bufB  = (float*)take((size_t)N * DCH * 4);    // hb1(bf16) / agg
    float*   dinv  = (float*)take((size_t)N * 4);
    __bf16*  wts   = (__bf16*)take(5 * DCH * DCH * 2);     // WT: enc1,enc2,conv0..2
    float*   stats = (float*)take(2 * DCH * 4);
    float*   scsh  = (float*)take(2 * DCH * 4);
    float*   hg    = (float*)take(64 * DCH * 4);

    const int GB = (N / 16 + 7) / 8;      // GEMM blocks (8 waves, 16 rows/wave)

    // --- degree -> dinv ---
    (void)hipMemsetAsync(dinv, 0, (size_t)N * 4, stream);
    k_deg<<<1024, 256, 0, stream>>>(ei + E, dinv, E);
    k_dinv<<<(N + 255) / 256, 256, 0, stream>>>(dinv, N);

    // --- weight transpose + bf16 ---
    k_wt<<<64, 256, 0, stream>>>(eW1, wts + 0 * DCH * DCH);
    k_wt<<<64, 256, 0, stream>>>(eW2, wts + 1 * DCH * DCH);
    for (int l = 0; l < 3; ++l)
        k_wt<<<64, 256, 0, stream>>>(cW + (size_t)l * DCH * DCH,
                                     wts + (2 + l) * DCH * DCH);

    // --- encoder ---
    k_bf16cast<<<4096, 256, 0, stream>>>(x, hb, n4);                 // x -> bf16
    (void)hipMemsetAsync(stats, 0, 2 * DCH * 4, stream);
    k_gemm_wmma<true, false><<<GB, 256, 0, stream>>>(
        hb, wts, eb1, bufA, nullptr, stats, N);                      // t = x@W1+b1 (+stats)
    k_finalize<<<1, DCH, 0, stream>>>(stats, eg, ebe, scsh, 1.0f / (float)N);
    k_bnrelu<<<4096, 256, 0, stream>>>(bufA, scsh, (__bf16*)bufB, n4);
    k_gemm_wmma<false, true><<<GB, 256, 0, stream>>>(
        (__bf16*)bufB, wts + DCH * DCH, eb2, h, hb, nullptr, N);     // h = relu_bn@W2+b2

    // --- GCN layers ---
    for (int l = 0; l < 3; ++l) {
        k_gemm_wmma<false, false><<<GB, 256, 0, stream>>>(
            hb, wts + (2 + l) * DCH * DCH, nullptr, bufA, nullptr, nullptr, N);
        (void)hipMemsetAsync(bufB, 0, (size_t)N * DCH * 4, stream);
        (void)hipMemsetAsync(stats, 0, 2 * DCH * 4, stream);
        k_scatter<<<2048, 256, 0, stream>>>(bufA, ei, ei + E, dinv, bufB, E);
        k_colstats<<<512, 256, 0, stream>>>(bufB, stats, N);
        k_finalize<<<1, DCH, 0, stream>>>(stats, bng + l * DCH, bnb + l * DCH,
                                          scsh, 1.0f / (float)N);
        k_update<<<4096, 256, 0, stream>>>(h, hb, bufB, scsh, n4);
    }

    // --- pooling + readout ---
    (void)hipMemsetAsync(hg, 0, 64 * DCH * 4, stream);
    k_pool<<<4096, 256, 0, stream>>>(h, batc, hg, N);
    k_readout<<<1, 256, 0, stream>>>(hg, rW1, rb1, rW2, rb2, rW3, rb3,
                                     (float*)d_out);
}
